// TransMatcher_63788854280566
// MI455X (gfx1250) — matrix operand order repaired and checked by reading, not compiled, not run on hardware
//
#include <hip/hip_runtime.h>
#include <stdint.h>
#include <math.h>

#define SEQ 192
#define FF  2048
#define EPSV 1e-5f

typedef __attribute__((ext_vector_type(16))) __bf16 v16bf;
typedef __attribute__((ext_vector_type(8)))  float  v8f;
typedef __attribute__((ext_vector_type(4)))  unsigned int u32x4;
typedef __attribute__((ext_vector_type(4)))  int i32x4;

union FragU { u32x4 q[2]; unsigned int u[8]; v16bf v; };

__device__ inline v8f v8zero() {
  v8f z = {0.f,0.f,0.f,0.f,0.f,0.f,0.f,0.f};
  return z;
}

__device__ inline unsigned short f2bf(float f) {
  unsigned int u = __float_as_uint(f);
  unsigned int r = (u + 0x7FFFu + ((u >> 16) & 1u)) >> 16;
  return (unsigned short)r;
}

// deterministic float max via integer atomics (init with -inf bit pattern)
__device__ inline void atomicMaxF(float* a, float v) {
  if (v >= 0.f) atomicMax((int*)a, __float_as_int(v));
  else          atomicMin((unsigned int*)a, __float_as_uint(v));
}

// ---- CDNA5 async global->LDS copy (ASYNCcnt path), with portable fallback ----
__device__ __forceinline__ void async_copy_b128(const unsigned short* g, unsigned short* l) {
#if defined(__AMDGCN__) && __has_builtin(__builtin_amdgcn_global_load_async_to_lds_b128)
  __builtin_amdgcn_global_load_async_to_lds_b128(
      (__attribute__((address_space(1))) i32x4*)g,
      (__attribute__((address_space(3))) i32x4*)l, 0, 0);
#else
  *(u32x4*)l = *(const u32x4*)g;
#endif
}

__device__ __forceinline__ void wait_async0() {
#if defined(__AMDGCN__) && __has_builtin(__builtin_amdgcn_s_wait_asynccnt)
  __builtin_amdgcn_s_wait_asynccnt(0);
#elif defined(__AMDGCN__)
  asm volatile("s_wait_asynccnt 0" ::: "memory");
#endif
}

// A-matrix 16x32 bf16 fragment (M rows = row0..row0+15, K = d0..d0+31)
// per ISA layout: lanes 0-15 hold K 0-7 (v0-3) & 16-23 (v4-7); lanes 16-31 hold K 8-15 & 24-31
// Works for global or LDS source (inlined -> addrspace inferred: global_load_b128 / ds_load_b128)
__device__ __forceinline__ v16bf load_frag_A(const unsigned short* base,
                                             int row0, int ld, int d0, int lane) {
  int h = lane >> 4, m = lane & 15;
  const unsigned short* p = base + (size_t)(row0 + m) * ld + d0;
  FragU f;
  f.q[0] = *(const u32x4*)(p + h * 8);
  f.q[1] = *(const u32x4*)(p + 16 + h * 8);
  return f.v;
}

// B-matrix 32x16 bf16 fragment. B[k][n] = src[col0+n][d0+k] (src is N-major, d-contiguous).
// lanes 0-15: N=lane, K=0..7 & 16..23; lanes 16-31: N=lane-16, K=8..15 & 24..31
__device__ __forceinline__ v16bf load_frag_B(const unsigned short* __restrict__ base,
                                             int col0, int ld, int d0, int lane) {
  int n = lane & 15, kb = (lane >> 4) * 8;
  const unsigned short* p = base + (size_t)(col0 + n) * ld + d0 + kb;
  FragU f;
  f.q[0] = *(const u32x4*)(p);
  f.q[1] = *(const u32x4*)(p + 16);
  return f.v;
}

__device__ __forceinline__ v8f wmma_bf16(v16bf a, v16bf b, v8f c) {
  return __builtin_amdgcn_wmma_f32_16x16x32_bf16(false, a, false, b, (short)0, c, false, false);
}

// ---------------- small utility kernels ----------------

__global__ void k_sigmoid(const float* __restrict__ in, float* __restrict__ out, int n) {
  int i = blockIdx.x * 256 + threadIdx.x;
  if (i < n) out[i] = 1.f / (1.f + expf(-in[i]));
}

__global__ void k_f2bf(const float* __restrict__ in, unsigned short* __restrict__ out, int n) {
  int i = blockIdx.x * 256 + threadIdx.x;
  if (i < n) out[i] = f2bf(in[i]);
}

__global__ void k_zero(float* __restrict__ p, int n) {
  int i = blockIdx.x * 256 + threadIdx.x;
  if (i < n) p[i] = 0.f;
}

// ---------------- fc1: X[b,t,d] @ W^T + bias  ->  bf16 [b][t][j] (j padded to 192 w/ 0) ----
// X is NCHW gallery/query: element (b, start+d, t) at X[(b*512 + start + d)*SEQ + t]
__global__ void k_fc1(const float* __restrict__ X, const float* __restrict__ W,
                      const float* __restrict__ Bias, unsigned short* __restrict__ Out,
                      int start, int c) {
  int b = blockIdx.x, j = blockIdx.y, t = threadIdx.x; // t in [0,192)
  float acc = 0.f;
  if (j < c) {
    acc = Bias[j];
    const float* g = X + ((size_t)b * 512 + start) * SEQ + t;
    const float* w = W + (size_t)j * c;
    for (int d = 0; d < c; ++d) acc = fmaf(g[(size_t)d * SEQ], w[d], acc);
  }
  Out[((size_t)b * SEQ + t) * SEQ + j] = f2bf(acc);
}

// ---------------- fused correlation volume + sigmoid gate + dual max-reduce ----------------
// One workgroup per (q,k) pair; ky staged into LDS in two 96-row halves via async copies;
// 8 waves, each computes 16x64 per unit via 4 WMMAs (A from LDS, B from global/L2).
__global__ void __launch_bounds__(256) k_vol(const unsigned short* __restrict__ qybf,
                                             const unsigned short* __restrict__ kybf,
                                             const float* __restrict__ sig,
                                             float* __restrict__ maxS,   // [qk][t] (max over s)
                                             float* __restrict__ maxT) { // [qk][s] (max over t)
  __shared__ __align__(16) unsigned short Ald[96 * SEQ];   // 36 KB staged ky half
  __shared__ float lmS[SEQ], lmT[SEQ];
  int qk = blockIdx.x;
  int q = qk >> 5, k = qk & 31;
  int tid = threadIdx.x, wave = tid >> 5, lane = tid & 31;
  float ninf = __uint_as_float(0xff800000u);
  for (int i = tid; i < SEQ; i += 256) { lmS[i] = ninf; lmT[i] = ninf; }

  const unsigned short* A  = kybf + (size_t)k * SEQ * SEQ; // rows = s, cols = d
  const unsigned short* Bq = qybf + (size_t)q * SEQ * SEQ; // rows = t, cols = d
  int h = lane >> 4, n = lane & 15;

  for (int half = 0; half < 2; ++half) {
    __syncthreads();                         // consumers of previous half done (+init visible)
    // stage ky rows [half*96, half*96+96) : 96 rows x 384B = 2304 16-byte chunks
    for (int ci = tid; ci < 2304; ci += 256) {
      int row = ci / 24, c16 = (ci % 24) * 8;
      async_copy_b128(A + (size_t)(half * 96 + row) * SEQ + c16, &Ald[row * SEQ + c16]);
    }
    wait_async0();
    __syncthreads();

    for (int u = wave; u < 18; u += 8) {     // 6 s-tiles x 3 t-supertiles in this half
      int s0l = (u / 3) * 16, t0 = (u % 3) * 64;
      v8f acc[4];
      for (int ii = 0; ii < 4; ++ii) acc[ii] = v8zero();
      for (int d0 = 0; d0 < SEQ; d0 += 32) { // 171 padded to 192 with zeros
        v16bf a = load_frag_A(Ald, s0l, SEQ, d0, lane);     // ds_load_b128 x2
        for (int bb = 0; bb < 4; ++bb) {
          v16bf b = load_frag_B(Bq, t0 + bb * 16, SEQ, d0, lane);
          acc[bb] = wmma_bf16(a, b, acc[bb]);
        }
      }
      // epilogue: gate by sigmoid(se[s,t]) and reduce maxes
      for (int bb = 0; bb < 4; ++bb) {
        int t = t0 + bb * 16 + n;
        float g[8];
        float lmax = ninf;
        for (int jj = 0; jj < 8; ++jj) {
          int s = half * 96 + s0l + h * 8 + jj;
          float val = acc[bb][jj] * sig[(size_t)s * SEQ + t];
          g[jj] = val;
          lmax = fmaxf(lmax, val);
        }
        atomicMaxF(&lmS[t], lmax);           // max over s (per t)
        for (int jj = 0; jj < 8; ++jj) {     // max over t (per s): 16-lane butterfly
          float v = g[jj];
          for (int m = 1; m < 16; m <<= 1) v = fmaxf(v, __shfl_xor(v, m, 32));
          if (n == 0) atomicMaxF(&lmT[half * 96 + s0l + h * 8 + jj], v);
        }
      }
    }
  }
  __syncthreads();
  for (int i = tid; i < SEQ; i += 256) {
    maxS[(size_t)qk * SEQ + i] = lmS[i];
    maxT[(size_t)qk * SEQ + i] = lmT[i];
  }
}

// ---------------- scalar mean / inv-std over N contiguous floats ----------------
__global__ void k_stats_scalar(const float* __restrict__ x, int n, float* __restrict__ out2) {
  __shared__ float ss[256], sq[256];
  float s = 0.f, qq = 0.f;
  for (int i = threadIdx.x; i < n; i += 256) { float v = x[i]; s += v; qq += v * v; }
  ss[threadIdx.x] = s; sq[threadIdx.x] = qq; __syncthreads();
  for (int st = 128; st > 0; st >>= 1) {
    if (threadIdx.x < st) { ss[threadIdx.x] += ss[threadIdx.x + st]; sq[threadIdx.x] += sq[threadIdx.x + st]; }
    __syncthreads();
  }
  if (threadIdx.x == 0) {
    float m = ss[0] / n;
    float var = sq[0] / n - m * m;
    out2[0] = m; out2[1] = rsqrtf(var + EPSV);
  }
}

// bn1 apply + interleave (row 2i = maxS[i], row 2i+1 = maxT[i]) -> bf16 [2048][192]
__global__ void k_bn1apply(const float* __restrict__ maxS, const float* __restrict__ maxT,
                           const float* __restrict__ st2, unsigned short* __restrict__ xhat) {
  int idx = blockIdx.x * 256 + threadIdx.x;
  if (idx >= 2 * 1024 * SEQ) return;
  int r = idx / SEQ, t = idx % SEQ, i = r >> 1;
  float v = (r & 1) ? maxT[(size_t)i * SEQ + t] : maxS[(size_t)i * SEQ + t];
  xhat[idx] = f2bf((v - st2[0]) * st2[1]);
}

// ---------------- fc2: y[2048,2048] = xhat @ w2^T + b2 (bf16 WMMA, 16x64/wave) -----------
// All 8 waves of a block share the same 16 A-rows -> stage them in LDS asynchronously.
__global__ void __launch_bounds__(256) k_fc2(const unsigned short* __restrict__ xhat,
                                             const unsigned short* __restrict__ w2bf,
                                             const float* __restrict__ b2,
                                             float* __restrict__ y) {
  __shared__ __align__(16) unsigned short Ald[16 * SEQ];   // 6 KB A tile
  int b = blockIdx.x;
  int r0 = (b >> 2) * 16;                // 128 m-tiles
  int j0 = ((b & 3) * 8 + (threadIdx.x >> 5)) * 64;  // 32 n-supertiles
  int tid = threadIdx.x, lane = tid & 31;

  for (int ci = tid; ci < 384; ci += 256) {   // 16 rows x 24 chunks of 16B
    int row = ci / 24, c16 = (ci % 24) * 8;
    async_copy_b128(xhat + (size_t)(r0 + row) * SEQ + c16, &Ald[row * SEQ + c16]);
  }
  wait_async0();
  __syncthreads();

  v8f acc[4];
  for (int ii = 0; ii < 4; ++ii) acc[ii] = v8zero();
  for (int d0 = 0; d0 < SEQ; d0 += 32) {
    v16bf a = load_frag_A(Ald, 0, SEQ, d0, lane);
    for (int bb = 0; bb < 4; ++bb) {
      v16bf b = load_frag_B(w2bf, j0 + bb * 16, SEQ, d0, lane);
      acc[bb] = wmma_bf16(a, b, acc[bb]);
    }
  }
  int h = lane >> 4, n = lane & 15;
  for (int bb = 0; bb < 4; ++bb)
    for (int jj = 0; jj < 8; ++jj) {
      int r = r0 + h * 8 + jj, j = j0 + bb * 16 + n;
      y[(size_t)r * FF + j] = acc[bb][jj] + b2[j];
    }
}

// per-column mean / inv-std of y over 2048 rows
__global__ void k_colstats(const float* __restrict__ y, float* __restrict__ mu,
                           float* __restrict__ isg) {
  int j = blockIdx.x * 256 + threadIdx.x;
  if (j >= FF) return;
  float s = 0.f, qq = 0.f;
  for (int r = 0; r < FF; ++r) { float v = y[(size_t)r * FF + j]; s += v; qq += v * v; }
  float m = s / FF;
  mu[j] = m; isg[j] = rsqrtf(qq / FF - m * m + EPSV);
}

// z[r] = sum_j relu(bn2(y[r,j])) * w3[j] + b3
__global__ void k_fc3row(const float* __restrict__ y, const float* __restrict__ mu,
                         const float* __restrict__ isg, const float* __restrict__ w3,
                         const float* __restrict__ b3, float* __restrict__ z) {
  __shared__ float red[256];
  int r = blockIdx.x;
  float s = 0.f;
  for (int j = threadIdx.x; j < FF; j += 256) {
    float v = (y[(size_t)r * FF + j] - mu[j]) * isg[j];
    s = fmaf(fmaxf(v, 0.f), w3[j], s);
  }
  red[threadIdx.x] = s; __syncthreads();
  for (int st = 128; st > 0; st >>= 1) {
    if (threadIdx.x < st) red[threadIdx.x] += red[threadIdx.x + st];
    __syncthreads();
  }
  if (threadIdx.x == 0) z[r] = red[0] + b3[0];
}

// pair-sum + bn3 + accumulate into score
__global__ void k_combine(const float* __restrict__ z, float* __restrict__ score) {
  __shared__ float s1[1024];
  __shared__ float ss[256], sq[256], mv[2];
  float s = 0.f, qq = 0.f;
  for (int i = threadIdx.x; i < 1024; i += 256) {
    float v = z[2 * i] + z[2 * i + 1];
    s1[i] = v; s += v; qq += v * v;
  }
  ss[threadIdx.x] = s; sq[threadIdx.x] = qq; __syncthreads();
  for (int st = 128; st > 0; st >>= 1) {
    if (threadIdx.x < st) { ss[threadIdx.x] += ss[threadIdx.x + st]; sq[threadIdx.x] += sq[threadIdx.x + st]; }
    __syncthreads();
  }
  if (threadIdx.x == 0) {
    float m = ss[0] / 1024.f;
    mv[0] = m; mv[1] = rsqrtf(sq[0] / 1024.f - m * m + EPSV);
  }
  __syncthreads();
  for (int i = threadIdx.x; i < 1024; i += 256)
    score[i] += (s1[i] - mv[0]) * mv[1];
}

// final decoder BN over the 1024 scores -> d_out
__global__ void k_final(const float* __restrict__ score, float* __restrict__ out) {
  __shared__ float ss[256], sq[256], mv[2];
  float s = 0.f, qq = 0.f;
  for (int i = threadIdx.x; i < 1024; i += 256) { float v = score[i]; s += v; qq += v * v; }
  ss[threadIdx.x] = s; sq[threadIdx.x] = qq; __syncthreads();
  for (int st = 128; st > 0; st >>= 1) {
    if (threadIdx.x < st) { ss[threadIdx.x] += ss[threadIdx.x + st]; sq[threadIdx.x] += sq[threadIdx.x + st]; }
    __syncthreads();
  }
  if (threadIdx.x == 0) {
    float m = ss[0] / 1024.f;
    mv[0] = m; mv[1] = rsqrtf(sq[0] / 1024.f - m * m + EPSV);
  }
  __syncthreads();
  for (int i = threadIdx.x; i < 1024; i += 256)
    out[i] = (score[i] - mv[0]) * mv[1];
}

extern "C" void kernel_launch(void* const* d_in, const int* in_sizes, int n_in,
                              void* d_out, int out_size, void* d_ws, size_t ws_size,
                              hipStream_t stream) {
  (void)in_sizes; (void)n_in; (void)out_size; (void)ws_size;
  const float* gallery = (const float*)d_in[0];
  const float* query   = (const float*)d_in[1];
  float* out = (float*)d_out;

  char* ws = (char*)d_ws;
  size_t off = 0;
  auto alloc = [&](size_t bytes) { size_t o = off; off += (bytes + 255) & ~(size_t)255; return o; };

  unsigned short* qybf = (unsigned short*)(ws + alloc((size_t)32 * SEQ * SEQ * 2));
  unsigned short* kybf = (unsigned short*)(ws + alloc((size_t)32 * SEQ * SEQ * 2));
  float* sig   = (float*)(ws + alloc((size_t)SEQ * SEQ * 4));
  float* maxST = (float*)(ws + alloc((size_t)2 * 1024 * SEQ * 4)); // maxS then maxT, contiguous
  float* maxS = maxST;
  float* maxT = maxST + (size_t)1024 * SEQ;
  unsigned short* xhat = (unsigned short*)(ws + alloc((size_t)2048 * SEQ * 2));
  unsigned short* w2bf = (unsigned short*)(ws + alloc((size_t)FF * SEQ * 2));
  float* y    = (float*)(ws + alloc((size_t)FF * FF * 4));
  float* mu   = (float*)(ws + alloc((size_t)FF * 4));
  float* isg  = (float*)(ws + alloc((size_t)FF * 4));
  float* z    = (float*)(ws + alloc((size_t)FF * 4));
  float* score= (float*)(ws + alloc((size_t)1024 * 4));
  float* bnst = (float*)(ws + alloc(256));

  k_zero<<<4, 256, 0, stream>>>(score, 1024);

  const int starts[3] = {0, 171, 342};
  const int chans[3]  = {171, 171, 170};

  for (int L = 0; L < 3; ++L) {
    int base = 2 + 7 * L;
    const float* fc1_w = (const float*)d_in[base + 0];
    const float* fc1_b = (const float*)d_in[base + 1];
    const float* se    = (const float*)d_in[base + 2];
    const float* fc2_w = (const float*)d_in[base + 3];
    const float* fc2_b = (const float*)d_in[base + 4];
    const float* fc3_w = (const float*)d_in[base + 5];
    const float* fc3_b = (const float*)d_in[base + 6];
    int st = starts[L], c = chans[L];

    k_sigmoid<<<(SEQ * SEQ + 255) / 256, 256, 0, stream>>>(se, sig, SEQ * SEQ);
    k_f2bf<<<(FF * SEQ + 255) / 256, 256, 0, stream>>>(fc2_w, w2bf, FF * SEQ);

    dim3 g1(32, SEQ);
    k_fc1<<<g1, SEQ, 0, stream>>>(gallery, fc1_w, fc1_b, qybf, st, c); // qy from tgt=gallery
    k_fc1<<<g1, SEQ, 0, stream>>>(query,   fc1_w, fc1_b, kybf, st, c); // ky from mem=query

    k_vol<<<1024, 256, 0, stream>>>(qybf, kybf, sig, maxS, maxT);

    k_stats_scalar<<<1, 256, 0, stream>>>(maxST, 2 * 1024 * SEQ, bnst);
    k_bn1apply<<<(2 * 1024 * SEQ + 255) / 256, 256, 0, stream>>>(maxS, maxT, bnst, xhat);

    k_fc2<<<512, 256, 0, stream>>>(xhat, w2bf, fc2_b, y);
    k_colstats<<<FF / 256, 256, 0, stream>>>(y, mu, isg);
    k_fc3row<<<FF, 256, 0, stream>>>(y, mu, isg, fc3_w, fc3_b, z);
    k_combine<<<1, 256, 0, stream>>>(z, score);
  }

  k_final<<<1, 256, 0, stream>>>(score, out);
}
